// Attention_60370060312683
// MI455X (gfx1250) — matrix operand order repaired and checked by reading, not compiled, not run on hardware
//
#include <hip/hip_runtime.h>
#include <hip/hip_bf16.h>

// ---------------------------------------------------------------------------
// CDNA5 (gfx1250, wave32) types
// ---------------------------------------------------------------------------
typedef __attribute__((ext_vector_type(16))) __bf16 v16bf;
typedef __attribute__((ext_vector_type(8)))  float  v8f;
typedef unsigned int u32;
typedef u32 __attribute__((ext_vector_type(4))) v4u;
typedef int __attribute__((ext_vector_type(4))) v4i;
typedef int __attribute__((ext_vector_type(8))) v8i;

__device__ __forceinline__ v8f wmma_bf16(v16bf a, v16bf b, v8f c) {
  return __builtin_amdgcn_wmma_f32_16x16x32_bf16(false, a, false, b,
                                                 (short)0, c, false, false);
}

// ---------------------------------------------------------------------------
// Tensor Data Mover: 1-D contiguous copy global -> LDS (nbytes multiple of 8).
// D# built per CDNA5 ISA 8.3/8.4: group0 = {count=1, lds_addr, global_addr,
// type=2}; group1 = {data_size=8B, tensor/tile dim0 = nbytes/8}.
// Issue from ONE wave; completion via s_wait_tensorcnt + workgroup barrier.
// ---------------------------------------------------------------------------
__device__ __forceinline__ void tdm_load_1d(u32 lds_off, const void* gsrc, u32 nbytes) {
  const unsigned long long ga = (unsigned long long)(size_t)gsrc;
  const u32 ne = nbytes >> 3;   // 8-byte elements
  v4u g0 = { 1u,                                   // count=1 (user descriptor)
             lds_off,                              // lds_addr (bytes)
             (u32)(ga & 0xFFFFFFFFu),              // global_addr[31:0]
             (u32)((ga >> 32) & 0x01FFFFFFu) | 0x80000000u };  // addr[56:32]|type=2
  v8i g1 = { (int)0x00030000u,                     // mask=0, data_size=3 (8B)
             (int)((ne & 0xFFFFu) << 16),          // tensor_dim0[15:0]
             (int)(((ne >> 16) & 0xFFFFu) | (1u << 16)), // td0[31:16] | tensor_dim1=1
             (int)((ne & 0xFFFFu) << 16),          // td1 hi=0 | tile_dim0
             1,                                    // tile_dim1=1, tile_dim2=0
             (int)ne,                              // tensor_dim0_stride[31:0]
             (int)((ne & 0xFFFFu) << 16),          // s0 hi=0 | tensor_dim1_stride lo
             (int)((ne >> 16) & 0xFFFFu) };        // tensor_dim1_stride hi
  v4i g2 = {0, 0, 0, 0};
  v4i g3 = {0, 0, 0, 0};
#if defined(__clang_major__) && (__clang_major__ >= 23)
  v8i g4 = {0, 0, 0, 0, 0, 0, 0, 0};
  __builtin_amdgcn_tensor_load_to_lds(g0, g1, g2, g3, g4, 0);
#else
  __builtin_amdgcn_tensor_load_to_lds(g0, g1, g2, g3, 0);
#endif
}

// ---------------------------------------------------------------------------
// Fragment loads (k = 16*half + s per lane, same order in A and B). frag_a / frag_bt are contiguous
// 2 x 16B per lane (lower to ds_load_b128 pairs).
// ---------------------------------------------------------------------------
__device__ __forceinline__ v16bf frag_a(const __bf16* s, int ld, int r0, int k0, int lane) {
  const int half = (lane >> 4) & 1, m = lane & 15;
  const __bf16* row = s + (size_t)(r0 + m) * ld + k0 + 16 * half;
  v16bf f;
#pragma unroll
  for (int v = 0; v < 8; ++v) {
    const int kb = 2 * v;
    f[2 * v]     = row[kb];
    f[2 * v + 1] = row[kb + 1];
  }
  return f;
}

// B fragment from transposed (n-major) source: B[k][n] = s[(n0+n)*ld + k0+k]
__device__ __forceinline__ v16bf frag_bt(const __bf16* s, int ld, int n0, int k0, int lane) {
  const int half = (lane >> 4) & 1, n = lane & 15;
  const __bf16* row = s + (size_t)(n0 + n) * ld + k0 + 16 * half;
  v16bf f;
#pragma unroll
  for (int v = 0; v < 8; ++v) {
    f[2 * v]     = row[2 * v];
    f[2 * v + 1] = row[2 * v + 1];
  }
  return f;
}

// B fragment from fragment-native global memory: each lane's 16 bf16 are
// contiguous -> one 32B load (2 x global_load_b128).
__device__ __forceinline__ v16bf frag_b_native(const __bf16* base, size_t blk, int lane) {
  return *(const v16bf*)(base + (blk * 32 + lane) * 16);
}

// ---------------------------------------------------------------------------
// One-time W swizzle: W[512,512] f32 -> bf16 B-fragment-native layout:
// out[(((n>>4)*16 + (k>>5))*32 + lane)*16 + pos],
//   lane = (n&15) | ((k>>4)&1)<<4, pos = k&15.
// ---------------------------------------------------------------------------
__global__ __launch_bounds__(256) void wswz_kernel(const float* __restrict__ W,
                                                   __bf16* __restrict__ O) {
  const int idx = blockIdx.x * 256 + threadIdx.x;   // 0..262143
  const int k = idx >> 9, n = idx & 511;
  const float w = W[idx];
  const int lane = (n & 15) | (((k >> 4) & 1) << 4);
  const size_t blk = (size_t)(n >> 4) * 16 + (k >> 5);
  O[(blk * 32 + lane) * 16 + (k & 15)] = (__bf16)w;
}

// ---------------------------------------------------------------------------
// Projection GEMM: C = A[M,512](f32) @ W(bf16, fragment-native) ; C in bf16.
// VMODE=0: row-major C.  VMODE=1: V-fragment-native C (per batch of 2048 rows:
// C[((b*64 + key>>5)*32 + d>>4)*32 + laneD]*16 + key&15).
// Block 256 (8 waves), tile 128x128, wave tile 64x32.
// ---------------------------------------------------------------------------
template <int VMODE>
__global__ __launch_bounds__(256) void proj_gemm_kernel(
    const float* __restrict__ A, const __bf16* __restrict__ Wsw,
    __bf16* __restrict__ C) {
  __shared__ __bf16 As[128 * 32];
  const int tid = threadIdx.x, lane = tid & 31, wv = tid >> 5;
  const int m0 = blockIdx.x * 128, n0 = blockIdx.y * 128;
  const int wr = wv >> 2, wc = wv & 3;

  v8f acc[4][2];
#pragma unroll
  for (int i = 0; i < 4; ++i)
#pragma unroll
    for (int j = 0; j < 2; ++j)
#pragma unroll
      for (int r = 0; r < 8; ++r) acc[i][j][r] = 0.f;

  for (int k0 = 0; k0 < 512; k0 += 32) {
    __syncthreads();
    // A tile 128x32 f32 -> bf16 LDS (row-major; frag_a reads are contiguous)
#pragma unroll
    for (int it = 0; it < 4; ++it) {
      const int idx = tid + it * 256;
      const int row = idx >> 3, kq = (idx & 7) << 2;
      const float4 v = *(const float4*)(A + (size_t)(m0 + row) * 512 + k0 + kq);
      As[row * 32 + kq + 0] = (__bf16)v.x;
      As[row * 32 + kq + 1] = (__bf16)v.y;
      As[row * 32 + kq + 2] = (__bf16)v.z;
      As[row * 32 + kq + 3] = (__bf16)v.w;
    }
    if (k0 + 32 < 512 && tid < 128)
      __builtin_prefetch(A + (size_t)(m0 + tid) * 512 + k0 + 32, 0, 3);
    __syncthreads();

    const int kblk = k0 >> 5;
    v16bf bf[2];
#pragma unroll
    for (int ci = 0; ci < 2; ++ci) {
      const size_t blk = (size_t)((n0 + wc * 32 + ci * 16) >> 4) * 16 + kblk;
      bf[ci] = frag_b_native(Wsw, blk, lane);
    }
#pragma unroll
    for (int ri = 0; ri < 4; ++ri) {
      const v16bf a = frag_a(As, 32, wr * 64 + ri * 16, 0, lane);
#pragma unroll
      for (int ci = 0; ci < 2; ++ci)
        acc[ri][ci] = wmma_bf16(a, bf[ci], acc[ri][ci]);
    }
  }

  const int half = lane >> 4, l16 = lane & 15;
#pragma unroll
  for (int ri = 0; ri < 4; ++ri)
#pragma unroll
    for (int ci = 0; ci < 2; ++ci)
#pragma unroll
      for (int r = 0; r < 8; ++r) {
        const int m = m0 + wr * 64 + ri * 16 + r + 8 * half;
        const int n = n0 + wc * 32 + ci * 16 + l16;
        if (VMODE == 0) {
          C[(size_t)m * 512 + n] = (__bf16)acc[ri][ci][r];
        } else {
          const int key = m & 2047, bb = m >> 11;
          const int laneD = (n & 15) | (((key >> 4) & 1) << 4);
          const size_t dst =
              ((((size_t)bb * 64 + (key >> 5)) * 32 + (n >> 4)) * 32 + laneD) * 16 +
              (key & 15);
          C[dst] = (__bf16)acc[ri][ci][r];
        }
      }
}

// ---------------------------------------------------------------------------
// Row LayerNorm in-place on bf16 rows of width 512. One wave per row.
// ---------------------------------------------------------------------------
__global__ __launch_bounds__(256) void ln_rows_kernel(
    __bf16* __restrict__ X, const float* __restrict__ g,
    const float* __restrict__ bta, int rows) {
  const int gw = (int)((blockIdx.x * blockDim.x + threadIdx.x) >> 5);
  const int lane = threadIdx.x & 31;
  if (gw >= rows) return;
  __bf16* row = X + (size_t)gw * 512;

  float xv[16];
  float s = 0.f;
#pragma unroll
  for (int i = 0; i < 16; ++i) { xv[i] = (float)row[lane + 32 * i]; s += xv[i]; }
#pragma unroll
  for (int o = 16; o; o >>= 1) s += __shfl_xor(s, o, 32);
  const float mu = s * (1.f / 512.f);
  float vs = 0.f;
#pragma unroll
  for (int i = 0; i < 16; ++i) { const float d = xv[i] - mu; vs += d * d; }
#pragma unroll
  for (int o = 16; o; o >>= 1) vs += __shfl_xor(vs, o, 32);
  const float rstd = rsqrtf(vs * (1.f / 512.f) + 1e-5f);
#pragma unroll
  for (int i = 0; i < 16; ++i) {
    const int c = lane + 32 * i;
    row[c] = (__bf16)((xv[i] - mu) * rstd * g[c] + bta[c]);
  }
}

// ---------------------------------------------------------------------------
// Flash attention + fused output LayerNorm.
// Grid (Sq/64, B), block 256 (8 waves).
// LDS: Qs 64KB | Ks 64KB | Ss 16KB | Ps 8KB | stats; Xs overlays Qs+Ks.
// Q/K tiles arrive via TDM (tensor_load_to_lds). V fragments are read
// directly from global in fragment-native layout (produced by V GEMM).
// ---------------------------------------------------------------------------
#define ATTN_LDS_BYTES (156416)

__global__ __launch_bounds__(256) void flash_attn_kernel(
    const __bf16* __restrict__ Q, const __bf16* __restrict__ K,
    const __bf16* __restrict__ Vsw, const float* __restrict__ ques,
    const unsigned char* __restrict__ key_mask,
    const float* __restrict__ g, const float* __restrict__ bt,
    float* __restrict__ out) {
  extern __shared__ char smem[];
  __bf16* Qs  = (__bf16*)smem;                     //  65536 B @ 0
  __bf16* Ks  = (__bf16*)(smem + 65536);           //  65536 B
  float*  Xs  = (float*)smem;                      // 131072 B overlay (epilogue)
  float*  Ss  = (float*)(smem + 131072);           //  16384 B
  __bf16* Ps  = (__bf16*)(smem + 147456);          //   8192 B
  float*  mrow = (float*)(smem + 155648);
  float*  lrow = mrow + 64;
  float*  arow = lrow + 64;

  const int tid = threadIdx.x, lane = tid & 31, wv = tid >> 5;
  const int half = lane >> 4, l16 = lane & 15;
  const int b = blockIdx.y;
  const int q0 = blockIdx.x * 64;
  const unsigned char* kmask = key_mask + (size_t)b * 2048;
  const float scale = 0.044194173824159216f;   // 1/sqrt(512)

  const __bf16* Qg = Q + ((size_t)b * 2048 + q0) * 512;
  const __bf16* Kg = K + (size_t)b * 2048 * 512;
  const __bf16* Vb = Vsw + ((size_t)b << 20);  // 2048*512 elems per batch

  if (tid < 64) { mrow[tid] = -INFINITY; lrow[tid] = 0.f; }

  if (wv == 0) tdm_load_1d(0u, Qg, 65536u);   // Q tile via TDM

  v8f acc[4][4];
#pragma unroll
  for (int i = 0; i < 4; ++i)
#pragma unroll
    for (int j = 0; j < 4; ++j)
#pragma unroll
      for (int r = 0; r < 8; ++r) acc[i][j][r] = 0.f;

  const int ntiles = blockIdx.x + 1;   // causal: key tiles with k0 <= q0
  for (int t = 0; t < ntiles; ++t) {
    const int k0 = t * 64;
    // --- K tile via TDM (wave 0 issues, waits, then all waves sync) ---
    if (wv == 0) {
      tdm_load_1d(65536u, Kg + (size_t)k0 * 512, 65536u);
      __builtin_amdgcn_s_wait_tensorcnt(0);
    }
    if (t + 1 < ntiles)   // prefetch next K tile (256B/lane covers 64KB)
      __builtin_prefetch((const char*)(Kg + (size_t)(k0 + 64) * 512) + tid * 256, 0, 3);
    __syncthreads();

    // --- S = Q @ K^T : wave -> row block rb, two 16-col key blocks ---
    {
      const int rb = wv >> 1;
      const int cb0 = (wv & 1) * 2;
      v8f s0, s1;
#pragma unroll
      for (int r = 0; r < 8; ++r) { s0[r] = 0.f; s1[r] = 0.f; }
      for (int kk = 0; kk < 512; kk += 32) {
        const v16bf a  = frag_a(Qs, 512, rb * 16, kk, lane);
        const v16bf b0 = frag_bt(Ks, 512, cb0 * 16, kk, lane);
        const v16bf b1 = frag_bt(Ks, 512, cb0 * 16 + 16, kk, lane);
        s0 = wmma_bf16(a, b0, s0);
        s1 = wmma_bf16(a, b1, s1);
      }
#pragma unroll
      for (int r = 0; r < 8; ++r) {
        const int m = rb * 16 + r + 8 * half;
        Ss[m * 64 + cb0 * 16 + l16]      = s0[r];
        Ss[m * 64 + cb0 * 16 + 16 + l16] = s1[r];
      }
    }
    __syncthreads();

    // --- online softmax (threads 0..63, one query row each) ---
    if (tid < 64) {
      const int qidx = q0 + tid;
      const float m_old = mrow[tid];
      float mx = m_old;
      for (int j = 0; j < 64; ++j) {
        const int kidx = k0 + j;
        float s = Ss[tid * 64 + j] * scale;
        if (kidx > qidx || kmask[kidx]) s = -INFINITY;
        Ss[tid * 64 + j] = s;
        mx = fmaxf(mx, s);
      }
      const float alpha = __expf(m_old - mx);
      float lsum = 0.f;
      for (int j = 0; j < 64; ++j) {
        const float p = __expf(Ss[tid * 64 + j] - mx);
        Ps[tid * 64 + j] = (__bf16)p;
        lsum += p;
      }
      mrow[tid] = mx;
      lrow[tid] = lrow[tid] * alpha + lsum;
      arow[tid] = alpha;
    }
    __syncthreads();

    // --- rescale accumulators, then O += P @ V (V frags direct from L2) ---
#pragma unroll
    for (int ri = 0; ri < 4; ++ri)
#pragma unroll
      for (int r = 0; r < 8; ++r) {
        const float al = arow[ri * 16 + r + 8 * half];
#pragma unroll
        for (int ci = 0; ci < 4; ++ci) acc[ri][ci][r] *= al;
      }
#pragma unroll
    for (int kk = 0; kk < 64; kk += 32) {
      const int keyblk = (k0 + kk) >> 5;
      v16bf pf[4], vf[4];
#pragma unroll
      for (int ri = 0; ri < 4; ++ri) pf[ri] = frag_a(Ps, 64, ri * 16, kk, lane);
#pragma unroll
      for (int ci = 0; ci < 4; ++ci)
        vf[ci] = frag_b_native(Vb, (size_t)keyblk * 32 + (wv * 4 + ci), lane);
#pragma unroll
      for (int ri = 0; ri < 4; ++ri)
#pragma unroll
        for (int ci = 0; ci < 4; ++ci)
          acc[ri][ci] = wmma_bf16(pf[ri], vf[ci], acc[ri][ci]);
    }
    __syncthreads();   // protect Ks/Ps for next iteration
  }

  // --- epilogue: x = O / l + residual; then row LayerNorm -> out ---
  const float* qres = ques + ((size_t)b * 2048 + q0) * 512;
#pragma unroll
  for (int ri = 0; ri < 4; ++ri)
#pragma unroll
    for (int ci = 0; ci < 4; ++ci)
#pragma unroll
      for (int r = 0; r < 8; ++r) {
        const int m = ri * 16 + r + 8 * half;
        const int n = wv * 64 + ci * 16 + l16;
        Xs[m * 512 + n] = acc[ri][ci][r] / lrow[m] + qres[(size_t)m * 512 + n];
      }
  __syncthreads();

  for (int rr = 0; rr < 8; ++rr) {
    const int m = wv * 8 + rr;
    float xv[16];
    float s = 0.f;
#pragma unroll
    for (int i = 0; i < 16; ++i) { xv[i] = Xs[m * 512 + lane + 32 * i]; s += xv[i]; }
#pragma unroll
    for (int o = 16; o; o >>= 1) s += __shfl_xor(s, o, 32);
    const float mu = s * (1.f / 512.f);
    float vs = 0.f;
#pragma unroll
    for (int i = 0; i < 16; ++i) { const float d = xv[i] - mu; vs += d * d; }
#pragma unroll
    for (int o = 16; o; o >>= 1) vs += __shfl_xor(vs, o, 32);
    const float rstd = rsqrtf(vs * (1.f / 512.f) + 1e-5f);
    float* orow = out + ((size_t)b * 2048 + q0 + m) * 512;
#pragma unroll
    for (int i = 0; i < 16; ++i) {
      const int c = lane + 32 * i;
      orow[c] = (xv[i] - mu) * rstd * g[c] + bt[c];
    }
  }
}

// ---------------------------------------------------------------------------
// Launch pipeline
// ---------------------------------------------------------------------------
extern "C" void kernel_launch(void* const* d_in, const int* in_sizes, int n_in,
                              void* d_out, int out_size, void* d_ws, size_t ws_size,
                              hipStream_t stream) {
  (void)in_sizes; (void)n_in; (void)out_size; (void)ws_size;
  const float* vals = (const float*)d_in[0];
  const float* keys = (const float*)d_in[1];
  const float* ques = (const float*)d_in[2];
  // d_in[3] causal mask handled analytically
  const unsigned char* key_mask = (const unsigned char*)d_in[4];
  const float* Wv = (const float*)d_in[5];
  const float* Wk = (const float*)d_in[6];
  const float* Wq = (const float*)d_in[7];
  const float* ln_k_g = (const float*)d_in[8];
  const float* ln_k_b = (const float*)d_in[9];
  const float* ln_q_g = (const float*)d_in[10];
  const float* ln_q_b = (const float*)d_in[11];
  const float* ln_o_g = (const float*)d_in[12];
  const float* ln_o_b = (const float*)d_in[13];
  float* out = (float*)d_out;

  const size_t M = 8 * 2048;            // 16384 rows
  __bf16* vswz = (__bf16*)d_ws;         // [M,512] V fragment-native
  __bf16* kb   = vswz + M * 512;        // [M,512] row-major
  __bf16* qb   = kb + M * 512;          // [M,512] row-major
  __bf16* wv_s = qb + M * 512;          // 512x512 fragment-native weights
  __bf16* wk_s = wv_s + 512 * 512;
  __bf16* wq_s = wk_s + 512 * 512;

  wswz_kernel<<<1024, 256, 0, stream>>>(Wv, wv_s);
  wswz_kernel<<<1024, 256, 0, stream>>>(Wk, wk_s);
  wswz_kernel<<<1024, 256, 0, stream>>>(Wq, wq_s);

  const dim3 ggrid(128, 4), gblk(256);
  proj_gemm_kernel<1><<<ggrid, gblk, 0, stream>>>(vals, wv_s, vswz);
  proj_gemm_kernel<0><<<ggrid, gblk, 0, stream>>>(keys, wk_s, kb);
  proj_gemm_kernel<0><<<ggrid, gblk, 0, stream>>>(ques, wq_s, qb);

  ln_rows_kernel<<<2048, 256, 0, stream>>>(kb, ln_k_g, ln_k_b, (int)M);
  ln_rows_kernel<<<2048, 256, 0, stream>>>(qb, ln_q_g, ln_q_b, (int)M);

  flash_attn_kernel<<<dim3(32, 8), 256, ATTN_LDS_BYTES, stream>>>(
      qb, kb, vswz, ques, key_mask, ln_o_g, ln_o_b, out);
}
